// CustomTorchMHASelf_33526514712748
// MI455X (gfx1250) — matrix operand order repaired and checked by reading, not compiled, not run on hardware
//
#include <hip/hip_runtime.h>
#include <math.h>

#if defined(__has_builtin)
#if __has_builtin(__builtin_amdgcn_global_load_async_to_lds_b128)
#define HAVE_ASYNC_LDS 1
#endif
#endif

// ---------------- CDNA5 WMMA types ----------------
typedef __attribute__((ext_vector_type(16))) __bf16 v16bf;
typedef __attribute__((ext_vector_type(8)))  float  v8f;
typedef __attribute__((ext_vector_type(4)))  int    v4i;

// async-to-LDS builtin pointer types: (v4i AS1* src, v4i AS3* dst, imm offset, imm cpol)
typedef __attribute__((address_space(1))) v4i* g4_t;
typedef __attribute__((address_space(3))) v4i* l4_t;

union Frag {
    v16bf v;
    uint4 q[2];
    unsigned short u[16];
};

static __device__ inline v8f wmma_bf16(v16bf a, v16bf b, v8f c) {
    // (neg_a, A, neg_b, B, c_mod, C, reuse_a, reuse_b)
    return __builtin_amdgcn_wmma_f32_16x16x32_bf16(false, a, false, b, (short)0, c, false, false);
}

static __device__ inline v8f vzero8() {
    v8f z;
#pragma unroll
    for (int i = 0; i < 8; ++i) z[i] = 0.0f;
    return z;
}

static __device__ inline unsigned short f2bf(float f) {
    union { float f; unsigned u; } c; c.f = f;
    unsigned r = c.u + 0x7FFFu + ((c.u >> 16) & 1u);   // round-to-nearest-even
    return (unsigned short)(r >> 16);
}
static __device__ inline float bf2f(unsigned short h) {
    union { unsigned u; float f; } c; c.u = ((unsigned)h) << 16;
    return c.f;
}

// Problem constants
#define BATCH 2
#define SEQ   2048
#define EMB   2048
#define HEADS 16
#define HDIM  128
#define ROWS  (BATCH * SEQ)          // 4096
#define N_QKV (3 * EMB)              // 6144

// LDS B-tile: 64 rows x 32 bf16, padded to 40 ushort (80B) row stride -> bank-conflict-free
#define SB_STRIDE 40
#define SB_ELEMS  (64 * SB_STRIDE)

// ---------------- f32 -> bf16 conversion ----------------
__global__ void f32_to_bf16_kernel(const float* __restrict__ src,
                                   unsigned short* __restrict__ dst, int n) {
    int i = blockIdx.x * 256 + threadIdx.x;
    if (i < n) dst[i] = f2bf(src[i]);
}

// ---------------- Block-cooperative 128x64 GEMM (4 waves x 32 rows), K chunks of 32 ----
// A: [M, lda] bf16 row-major (per-wave direct global loads).
// B: [N, ldb] bf16 row-major; the shared 64x32 tile is staged in LDS once per block,
// double-buffered via gfx1250 async-to-LDS copies (ASYNCcnt) when available.
static __device__ inline void gemm_block_32x64(const unsigned short* __restrict__ A, int lda,
                                               const unsigned short* __restrict__ B, int ldb,
                                               int K, int row0, int col0,
                                               unsigned short* sB, v8f acc[2][4]) {
    const int tid  = threadIdx.x;
    const int lane = tid & 31;
    const int half = lane >> 4;
    const int l16  = lane & 15;
    // staging assignment: thread -> (row 0..63, 32B half 0..1) of the 64x64B tile
    const int srow = tid >> 1;
    const int sh2  = tid & 1;
    const unsigned short* gB0 = B + (size_t)(col0 + srow) * ldb + sh2 * 16;
    unsigned short* lB0 = sB + srow * SB_STRIDE + sh2 * 16;

    auto stage = [&](int chunk, int buf) {
        const unsigned short* g = gB0 + chunk * 32;
        unsigned short* l = lB0 + buf * SB_ELEMS;
#ifdef HAVE_ASYNC_LDS
        __builtin_amdgcn_global_load_async_to_lds_b128((g4_t)g, (l4_t)l, 0, 0);
        __builtin_amdgcn_global_load_async_to_lds_b128((g4_t)(g + 8), (l4_t)(l + 8), 0, 0);
#else
        uint4 t0 = *(const uint4*)(g);
        uint4 t1 = *(const uint4*)(g + 8);
        *(uint4*)(l)     = t0;
        *(uint4*)(l + 8) = t1;
#endif
    };

    const int nchunks = K / 32;
    stage(0, 0);
    for (int i = 0; i < nchunks; ++i) {
        const int buf = i & 1;
        const bool more = (i + 1) < nchunks;
        if (more) stage(i + 1, buf ^ 1);
#ifdef HAVE_ASYNC_LDS
        // async loads complete in order: <=2 outstanding means chunk i has landed
        if (more) asm volatile("s_wait_asynccnt 0x2" ::: "memory");
        else      asm volatile("s_wait_asynccnt 0x0" ::: "memory");
#endif
        __syncthreads();   // everyone's chunk-i tile visible

        const int kc = i * 32;
        const unsigned short* sbuf = sB + buf * SB_ELEMS;
        Frag a[2], b[4];
#pragma unroll
        for (int mg = 0; mg < 2; ++mg) {
            const unsigned short* ap = A + (size_t)(row0 + mg * 16 + l16) * lda + kc + 16 * half;
            a[mg].q[0] = *(const uint4*)(ap);        // K = kc+16h .. +7
            a[mg].q[1] = *(const uint4*)(ap + 8);    // K = kc+16h+8 .. +7
            if (more) __builtin_prefetch(ap + 32, 0, 3);
        }
#pragma unroll
        for (int ng = 0; ng < 4; ++ng) {
            const unsigned short* bp = sbuf + (ng * 16 + l16) * SB_STRIDE + 16 * half;
            b[ng].q[0] = *(const uint4*)(bp);        // K = kc+16h .. +7
            b[ng].q[1] = *(const uint4*)(bp + 8);
        }
#pragma unroll
        for (int mg = 0; mg < 2; ++mg)
#pragma unroll
            for (int ng = 0; ng < 4; ++ng)
                acc[mg][ng] = wmma_bf16(a[mg].v, b[ng].v, acc[mg][ng]);

        __syncthreads();   // done reading buf before it is overwritten (i+2)
    }
}

// ---------------- QKV projection: qkv = x @ Wqkv^T + b, scattered to Q/K/Vt ----------------
// Q,K: [B*H, S, D] bf16 (pre-RoPE).  Vt: [B*H, D, S] bf16 (transposed for PV B-frags).
__global__ __launch_bounds__(128)
void qkv_gemm_kernel(const unsigned short* __restrict__ xb,
                     const unsigned short* __restrict__ wb,
                     const float* __restrict__ bias,
                     unsigned short* __restrict__ Qr,
                     unsigned short* __restrict__ Kr,
                     unsigned short* __restrict__ Vt) {
    __shared__ __align__(16) unsigned short sB[2 * SB_ELEMS];
    const int wave = threadIdx.x >> 5;
    const int col0 = blockIdx.x * 64;
    const int row0 = blockIdx.y * 128 + wave * 32;
    const int lane = threadIdx.x & 31;
    const int half = lane >> 4;
    const int l16  = lane & 15;

    v8f acc[2][4];
#pragma unroll
    for (int i = 0; i < 2; ++i)
#pragma unroll
        for (int j = 0; j < 4; ++j) acc[i][j] = vzero8();

    gemm_block_32x64(xb, EMB, wb, EMB, EMB, row0, col0, sB, acc);

#pragma unroll
    for (int mg = 0; mg < 2; ++mg) {
#pragma unroll
        for (int ng = 0; ng < 4; ++ng) {
            const int n     = col0 + ng * 16 + l16;     // column in [0, 3E)
            const float bv  = bias[n];
            const int which = n >> 11;                  // / 2048 : 0=q 1=k 2=v
            const int rem   = n & (EMB - 1);
            const int h     = rem >> 7;                 // / 128
            const int d     = rem & (HDIM - 1);
#pragma unroll
            for (int r = 0; r < 8; ++r) {
                const int m = row0 + mg * 16 + r + 8 * half;   // row in [0, B*S)
                const int b = m >> 11;
                const int s = m & (SEQ - 1);
                const int bh = b * HEADS + h;
                const unsigned short val = f2bf(acc[mg][ng][r] + bv);
                if (which == 0)      Qr[((size_t)bh * SEQ + s) * HDIM + d] = val;
                else if (which == 1) Kr[((size_t)bh * SEQ + s) * HDIM + d] = val;
                else                 Vt[((size_t)bh * HDIM + d) * SEQ + s] = val;
            }
        }
    }
}

// ---------------- RoPE (ESM rotate_half) in-place on Q and K ----------------
__global__ __launch_bounds__(128)
void rope_qk_kernel(unsigned short* __restrict__ Qr, unsigned short* __restrict__ Kr) {
    const int gid = blockIdx.x * 2 + (threadIdx.x >> 6);   // row in [0, B*H*S)
    const int j   = threadIdx.x & 63;                      // d in [0,64), pairs with d+64
    const int s   = gid & (SEQ - 1);
    const float invf = __powf(10000.0f, -(float)j * (1.0f / 64.0f));
    float sn, cs;
    __sincosf((float)s * invf, &sn, &cs);
    const size_t base = (size_t)gid * HDIM;

    float q1 = bf2f(Qr[base + j]), q2 = bf2f(Qr[base + j + 64]);
    Qr[base + j]      = f2bf(q1 * cs - q2 * sn);
    Qr[base + j + 64] = f2bf(q2 * cs + q1 * sn);

    float k1 = bf2f(Kr[base + j]), k2 = bf2f(Kr[base + j + 64]);
    Kr[base + j]      = f2bf(k1 * cs - k2 * sn);
    Kr[base + j + 64] = f2bf(k2 * cs + k1 * sn);
}

// ---------------- Flash attention: 1 wave = 16 q-rows, online softmax ----------------
__global__ __launch_bounds__(128)
void attn_kernel(const unsigned short* __restrict__ Qr,
                 const unsigned short* __restrict__ Kr,
                 const unsigned short* __restrict__ Vt,
                 unsigned short* __restrict__ ctx) {
    __shared__ __align__(16) unsigned short sP[4][16 * 32];   // per-wave P bounce (C->A layout)

    const int bh   = blockIdx.x >> 5;          // 32 q-tiles of 64 rows per (b,h)
    const int qt   = blockIdx.x & 31;
    const int wave = threadIdx.x >> 5;
    const int qrow0 = qt * 64 + wave * 16;
    const int lane = threadIdx.x & 31;
    const int half = lane >> 4;
    const int l16  = lane & 15;

    const unsigned short* Qb = Qr + (size_t)bh * SEQ * HDIM;
    const unsigned short* Kb = Kr + (size_t)bh * SEQ * HDIM;
    const unsigned short* Vb = Vt + (size_t)bh * HDIM * SEQ;

    // Preload the 4 Q A-fragments (d-chunks of 32), reused for all 2048 keys.
    Frag qf[4];
#pragma unroll
    for (int c = 0; c < 4; ++c) {
        const unsigned short* ap = Qb + (size_t)(qrow0 + l16) * HDIM + c * 32 + 16 * half;
        qf[c].q[0] = *(const uint4*)(ap);
        qf[c].q[1] = *(const uint4*)(ap + 8);
    }

    v8f O[8];
#pragma unroll
    for (int g = 0; g < 8; ++g) O[g] = vzero8();
    float mrow[8], lrow[8];
#pragma unroll
    for (int r = 0; r < 8; ++r) { mrow[r] = -1e30f; lrow[r] = 0.0f; }

    const float scale = 0.0883883476483184f;   // 1/sqrt(128)
    unsigned short* myP = &sP[wave][0];

    for (int kc = 0; kc < SEQ; kc += 32) {
        // prefetch next chunk of K (rows) and V (transposed rows)
        if (kc + 32 < SEQ) {
            __builtin_prefetch(Kb + (size_t)(kc + 32 + lane) * HDIM, 0, 3);
            __builtin_prefetch(Vb + (size_t)(lane * 4) * SEQ + kc + 32, 0, 3);
        }
        // ---- S = Q K^T for 32 keys (two 16-key groups) ----
        v8f sacc[2];
        sacc[0] = vzero8(); sacc[1] = vzero8();
#pragma unroll
        for (int g = 0; g < 2; ++g) {
            const int key0 = kc + g * 16;
#pragma unroll
            for (int c = 0; c < 4; ++c) {
                Frag kf;
                const unsigned short* bp = Kb + (size_t)(key0 + l16) * HDIM + c * 32 + 16 * half;
                kf.q[0] = *(const uint4*)(bp);
                kf.q[1] = *(const uint4*)(bp + 8);
                sacc[g] = wmma_bf16(qf[c].v, kf.v, sacc[g]);
            }
        }
        // scale + chunk row max (reduce across the 16 lanes of this half)
        float rmax[8];
#pragma unroll
        for (int r = 0; r < 8; ++r) {
            sacc[0][r] *= scale;
            sacc[1][r] *= scale;
            rmax[r] = fmaxf(sacc[0][r], sacc[1][r]);
        }
#pragma unroll
        for (int mask = 1; mask < 16; mask <<= 1)
#pragma unroll
            for (int r = 0; r < 8; ++r)
                rmax[r] = fmaxf(rmax[r], __shfl_xor(rmax[r], mask, 32));

        // online-softmax update
        float alpha[8], psum[8];
#pragma unroll
        for (int r = 0; r < 8; ++r) {
            const float mnew = fmaxf(mrow[r], rmax[r]);
            alpha[r] = __expf(mrow[r] - mnew);
            mrow[r]  = mnew;
            psum[r]  = 0.0f;
        }
        // p = exp(s - m), write bf16 P tile (16x32) to LDS in (row, key) order
#pragma unroll
        for (int g = 0; g < 2; ++g)
#pragma unroll
            for (int r = 0; r < 8; ++r) {
                const float p = __expf(sacc[g][r] - mrow[r]);
                psum[r] += p;
                myP[(r + 8 * half) * 32 + g * 16 + l16] = f2bf(p);
            }
#pragma unroll
        for (int mask = 1; mask < 16; mask <<= 1)
#pragma unroll
            for (int r = 0; r < 8; ++r)
                psum[r] += __shfl_xor(psum[r], mask, 32);
#pragma unroll
        for (int r = 0; r < 8; ++r) lrow[r] = lrow[r] * alpha[r] + psum[r];

        // rescale running O by alpha (row-matched: O VGPR r == row r+8*half)
#pragma unroll
        for (int g = 0; g < 8; ++g)
#pragma unroll
            for (int r = 0; r < 8; ++r) O[g][r] *= alpha[r];

        // same-wave LDS RAW: make the P stores visible before A-frag loads
        asm volatile("s_wait_dscnt 0" ::: "memory");

        // ---- O += P V  (A = P from LDS, B = Vt rows, contiguous along S) ----
        Frag pf;
        {
            const unsigned short* pp = myP + l16 * 32 + 16 * half;
            pf.q[0] = *(const uint4*)(pp);        // keys kc+16h .. +7
            pf.q[1] = *(const uint4*)(pp + 8);    // keys kc+16h+8 .. +7
        }
#pragma unroll
        for (int g = 0; g < 8; ++g) {
            Frag vf;
            const unsigned short* vp = Vb + (size_t)(g * 16 + l16) * SEQ + kc + 16 * half;
            vf.q[0] = *(const uint4*)(vp);
            vf.q[1] = *(const uint4*)(vp + 8);
            O[g] = wmma_bf16(pf.v, vf.v, O[g]);
        }
        asm volatile("s_wait_dscnt 0" ::: "memory");   // reads done before next chunk's stores
    }

    // epilogue: O /= l, write ctx [B*S, E] bf16
    const int h = bh & (HEADS - 1);
    const int b = bh >> 4;
#pragma unroll
    for (int r = 0; r < 8; ++r) lrow[r] = 1.0f / lrow[r];
#pragma unroll
    for (int g = 0; g < 8; ++g) {
        const int col = h * HDIM + g * 16 + l16;
#pragma unroll
        for (int r = 0; r < 8; ++r) {
            const int s = qrow0 + r + 8 * half;
            ctx[((size_t)(b * SEQ + s)) * EMB + col] = f2bf(O[g][r] * lrow[r]);
        }
    }
}

// ---------------- Output projection: out = ctx @ out_w^T + b (fp32 out) ----------------
__global__ __launch_bounds__(128)
void out_gemm_kernel(const unsigned short* __restrict__ ctxb,
                     const unsigned short* __restrict__ wb,
                     const float* __restrict__ bias,
                     float* __restrict__ out) {
    __shared__ __align__(16) unsigned short sB[2 * SB_ELEMS];
    const int wave = threadIdx.x >> 5;
    const int col0 = blockIdx.x * 64;
    const int row0 = blockIdx.y * 128 + wave * 32;
    const int lane = threadIdx.x & 31;
    const int half = lane >> 4;
    const int l16  = lane & 15;

    v8f acc[2][4];
#pragma unroll
    for (int i = 0; i < 2; ++i)
#pragma unroll
        for (int j = 0; j < 4; ++j) acc[i][j] = vzero8();

    gemm_block_32x64(ctxb, EMB, wb, EMB, EMB, row0, col0, sB, acc);

#pragma unroll
    for (int mg = 0; mg < 2; ++mg)
#pragma unroll
        for (int ng = 0; ng < 4; ++ng) {
            const int n    = col0 + ng * 16 + l16;
            const float bv = bias[n];
#pragma unroll
            for (int r = 0; r < 8; ++r) {
                const int m = row0 + mg * 16 + r + 8 * half;
                out[(size_t)m * EMB + n] = acc[mg][ng][r] + bv;
            }
        }
}

// ---------------- host launch ----------------
extern "C" void kernel_launch(void* const* d_in, const int* in_sizes, int n_in,
                              void* d_out, int out_size, void* d_ws, size_t ws_size,
                              hipStream_t stream) {
    const float* x    = (const float*)d_in[0];   // [B,S,E]
    const float* wqkv = (const float*)d_in[1];   // [3E,E]
    const float* bqkv = (const float*)d_in[2];   // [3E]
    const float* wout = (const float*)d_in[3];   // [E,E]
    const float* bout = (const float*)d_in[4];   // [E]
    float* out = (float*)d_out;

    char* ws = (char*)d_ws;
    // workspace layout (bytes), total ~96 MiB
    unsigned short* xb    = (unsigned short*)(ws);              // 16 MiB: x bf16, reused as ctx
    unsigned short* wqkvb = (unsigned short*)(ws + 16777216);   // 24 MiB
    unsigned short* owb   = (unsigned short*)(ws + 41943040);   // 8 MiB
    unsigned short* Qr    = (unsigned short*)(ws + 50331648);   // 16 MiB [B*H,S,D]
    unsigned short* Kr    = (unsigned short*)(ws + 67108864);   // 16 MiB [B*H,S,D]
    unsigned short* Vt    = (unsigned short*)(ws + 83886080);   // 16 MiB [B*H,D,S]

    const int n_x = ROWS * EMB;          // 8388608
    const int n_w = N_QKV * EMB;         // 12582912
    const int n_o = EMB * EMB;           // 4194304
    f32_to_bf16_kernel<<<(n_x + 255) / 256, 256, 0, stream>>>(x, xb, n_x);
    f32_to_bf16_kernel<<<(n_w + 255) / 256, 256, 0, stream>>>(wqkv, wqkvb, n_w);
    f32_to_bf16_kernel<<<(n_o + 255) / 256, 256, 0, stream>>>(wout, owb, n_o);

    qkv_gemm_kernel<<<dim3(N_QKV / 64, ROWS / 128), 128, 0, stream>>>(xb, wqkvb, bqkv, Qr, Kr, Vt);
    rope_qk_kernel<<<BATCH * HEADS * SEQ / 2, 128, 0, stream>>>(Qr, Kr);
    attn_kernel<<<BATCH * HEADS * (SEQ / 64), 128, 0, stream>>>(Qr, Kr, Vt, xb);  // ctx -> xb
    out_gemm_kernel<<<dim3(EMB / 64, ROWS / 128), 128, 0, stream>>>(xb, owb, bout, out);
}
